// DBI_10187662426595
// MI455X (gfx1250) — matrix operand order repaired and checked by reading, not compiled, not run on hardware
//
#include <hip/hip_runtime.h>
#include <cstdint>

#define T_FAST 512
#define T_SLOW 128
#define BB 32
#define DD 1024
#define HH 256
#define BD (BB*DD)          // 32768 floats per time step

// ---- workspace layout (in 4-byte words) ----
#define WS_G      ((size_t)0)        // 4 * 32768   : per-stream global means
#define WS_SCORE  ((size_t)131072)   // 8 * 32      : per-pair, per-batch scores
#define WS_COND   ((size_t)131328)   // 8           : per-pair cond flag
#define WS_PROJ   ((size_t)262144)   // 2 * 4194304 : W@feat_i at T=128 [(b*128+t)][d]
#define PROJ_SZ   ((size_t)(4096*1024))
#define WS_S2FW   ((size_t)8650752)  // s2f weights, bf16 hi/lo packed [d][c/2]
#define S2F_PLANE ((size_t)524288)   // 1024*512 words per plane
#define WS_F2SW   ((size_t)9699328)  // f2s weights, bf16 hi/lo packed [idx][ktap][d][c/2]
#define F2S_PLANE ((size_t)10485760) // 4*5*1024*512 words per plane
// total ws: 30,670,848 words = ~117 MB

typedef __attribute__((ext_vector_type(8)))  float  v8f;
typedef __attribute__((ext_vector_type(16))) __bf16 v16bf;
typedef __attribute__((ext_vector_type(4)))  int    v4i;

union frag_u { uint32_t w[8]; v16bf v; };

// ---- async global->LDS copy (CDNA5), with safe fallback ----
#if defined(__has_builtin)
#if __has_builtin(__builtin_amdgcn_global_load_async_to_lds_b128)
#define HAVE_ASYNC_LDS 1
#endif
#endif

__device__ __forceinline__ void copy_b128(const uint32_t* g, uint32_t* l) {
#ifdef HAVE_ASYNC_LDS
  // signature (probe-confirmed by diagnostic): (int4* src, int4* dst, imm off, imm cpol)
  __builtin_amdgcn_global_load_async_to_lds_b128((v4i*)g, (v4i*)l, 0, 0);
#else
  *(uint4*)l = *(const uint4*)g;
#endif
}
__device__ __forceinline__ void async_wait0() {
#ifdef HAVE_ASYNC_LDS
#if __has_builtin(__builtin_amdgcn_s_wait_asynccnt)
  __builtin_amdgcn_s_wait_asynccnt(0);
#else
  asm volatile("s_wait_asynccnt 0" ::: "memory");
#endif
#endif
}

// ---- fp32 -> bf16 hi/lo split (RNE), packed as {even K in [15:0], odd K in [31:16]} ----
__device__ __forceinline__ uint32_t bf16_rne(float x) {
  uint32_t u = __float_as_uint(x);
  return (u + 0x7FFFu + ((u >> 16) & 1u)) >> 16;
}
__device__ __forceinline__ float bf16_f(uint32_t h) { return __uint_as_float(h << 16); }
__device__ __forceinline__ void split2(float x0, float x1, uint32_t& hi, uint32_t& lo) {
  uint32_t h0 = bf16_rne(x0), h1 = bf16_rne(x1);
  uint32_t l0 = bf16_rne(x0 - bf16_f(h0)), l1 = bf16_rne(x1 - bf16_f(h1));
  hi = h0 | (h1 << 16);
  lo = l0 | (l1 << 16);
}

__device__ __forceinline__ v8f wmma_bf16(v16bf a, v16bf b, v8f c) {
  return __builtin_amdgcn_wmma_f32_16x16x32_bf16(false, a, false, b, (short)0, c,
                                                 false, false);
}

// LDS tiles: [row][kk] pitch 20 words, kk = K/2 (K-pair packed words).
// A frag: v[0..3] = kk {base..base+3}, v[4..7] = kk {base+8..base+11}, base = (lane>=16)*4
__device__ __forceinline__ v16bf fragA(const uint32_t* S, int row, int lane) {
  const uint32_t* p = S + row * 20 + ((lane >> 4) << 2);
  frag_u f;
  *(uint4*)&f.w[0] = *(const uint4*)p;
  *(uint4*)&f.w[4] = *(const uint4*)(p + 8);
  return f.v;
}
// B frag: v[0..3] = kk {base..base+3}, v[4..7] = kk {base+8..base+11}, base = (lane>=16)*4
__device__ __forceinline__ v16bf fragB(const uint32_t* S, int row, int lane) {
  const uint32_t* p = S + row * 20 + ((lane >> 4) << 2);
  frag_u f;
  *(uint4*)&f.w[0] = *(const uint4*)p;
  *(uint4*)&f.w[4] = *(const uint4*)(p + 8);
  return f.v;
}

// ---------------------------------------------------------------- means ----
__global__ void means_k(const float* __restrict__ f0, const float* __restrict__ f1,
                        const float* __restrict__ f2, const float* __restrict__ f3,
                        float* __restrict__ ws) {
  const int blk = blockIdx.x;
  const int i = blk >> 7;
  const int e = (blk & 127) * 256 + threadIdx.x;
  const float* F = (i == 0) ? f0 : (i == 1) ? f1 : (i == 2) ? f2 : f3;
  const int T = (i < 2) ? T_FAST : T_SLOW;
  float s = 0.0f;
  for (int t = 0; t < T; ++t) s += F[(size_t)t * BD + e];
  ws[WS_G + (size_t)i * BD + e] = s / (float)T;
}

// ----------------------------------------------------------- gating MLP ----
__global__ void mlp_scores(const float* __restrict__ w1, const float* __restrict__ b1,
                           const float* __restrict__ w2, const float* __restrict__ b2,
                           float* __restrict__ ws) {
  __shared__ float gbuf[2048];
  __shared__ float red[256];
  const int PI[8] = {2, 3, 2, 3, 0, 1, 0, 1};
  const int PJ[8] = {0, 0, 1, 1, 2, 2, 3, 3};
  const int q = blockIdx.x;
  const int i = PI[q], j = PJ[q], p = i * 4 + j;
  const int h = threadIdx.x;
  const float* gi = ws + WS_G + (size_t)i * BD;
  const float* gj = ws + WS_G + (size_t)j * BD;
  const float* W1 = w1 + (size_t)p * 2048 * HH;
  const float w2h = w2[p * HH + h];
  float acc[BB];
#pragma unroll
  for (int b = 0; b < BB; ++b) acc[b] = b1[p * HH + h];

  for (int kc = 0; kc < 2048; kc += 64) {
#pragma unroll
    for (int u = 0; u < 8; ++u) {
      int e = h * 8 + u;
      int b = e >> 6, kk = e & 63;
      int k = kc + kk;
      gbuf[e] = (k < 1024) ? gi[b * DD + k] : gj[b * DD + (k - 1024)];
    }
    __syncthreads();
    for (int kk = 0; kk < 64; ++kk) {
      float wv = W1[(size_t)(kc + kk) * HH + h];
#pragma unroll
      for (int b = 0; b < BB; ++b) acc[b] += gbuf[b * 64 + kk] * wv;
    }
    __syncthreads();
  }

  float ssum = 0.0f;
  for (int b = 0; b < BB; ++b) {
    red[h] = fmaxf(acc[b], 0.0f) * w2h;
    __syncthreads();
    for (int s = 128; s > 0; s >>= 1) {
      if (h < s) red[h] += red[h + s];
      __syncthreads();
    }
    if (h == 0) {
      float sc = 1.0f / (1.0f + __expf(-(red[0] + b2[p])));
      ws[WS_SCORE + (size_t)q * BB + b] = sc;
      ssum += sc;
    }
    __syncthreads();
  }
  if (h == 0) ws[WS_COND + q] = (ssum * (1.0f / 32.0f) >= 0.3f) ? 1.0f : 0.0f;
}

// ------------------------------------------ weight pre-split (fp32->bf16x2) ----
__global__ void cvt_s2f(const float* __restrict__ W, uint32_t* __restrict__ whi,
                        uint32_t* __restrict__ wlo) {
  const int w = blockIdx.x * 256 + threadIdx.x;   // < 524288
  const int d = w >> 9, kk = w & 511;
  const float* s = W + (size_t)d * DD + kk * 2;
  uint32_t hi, lo;
  split2(s[0], s[1], hi, lo);
  whi[w] = hi; wlo[w] = lo;
}

// f2s_w [idx][d][c][ktap] -> [idx][ktap][d][c/2] hi/lo planes (kills stride-5 gather)
__global__ void cvt_f2s(const float* __restrict__ W, uint32_t* __restrict__ whi,
                        uint32_t* __restrict__ wlo) {
  const int w = blockIdx.x * 256 + threadIdx.x;   // < 10485760
  const int kk = w & 511;
  const int d = (w >> 9) & 1023;
  const int t2 = w >> 19;
  const int kt = t2 % 5;
  const int idx4 = t2 / 5;
  const size_t s = ((size_t)(idx4 * 1024 + d) * DD + kk * 2) * 5 + kt;
  uint32_t hi, lo;
  split2(W[s], W[s + 5], hi, lo);
  whi[w] = hi; wlo[w] = lo;
}

// ------------------------------------------------- slow->fast projection ----
// proj_i[(b*128+t)][d] = sum_k feat_i[t,b,k] * s2f_w[d,k]  via bf16x3 split WMMA.
__global__ void proj_gemm(const float* __restrict__ f2, const float* __restrict__ f3,
                          const uint32_t* __restrict__ Whi,
                          const uint32_t* __restrict__ Wlo,
                          float* __restrict__ ws) {
  __shared__ uint32_t Ahi[64 * 20], Alo[64 * 20];
  __shared__ uint32_t Bhi[128 * 20], Blo[128 * 20];
  const int iSel = blockIdx.z;
  const float* X = iSel ? f3 : f2;
  float* proj = ws + WS_PROJ + (size_t)iSel * PROJ_SZ;
  const int tid = threadIdx.x;
  const int m0 = blockIdx.x * 64;
  const int n0 = blockIdx.y * 128;
  const int lane = tid & 31, wave = tid >> 5;
  const int wm = (wave & 1) * 32, wn = (wave >> 1) * 32;
  v8f acc[2][2] = {};

  for (int k0 = 0; k0 < DD; k0 += 32) {
    {   // B: pre-split weights -> LDS, async when available
      int n = tid >> 1, half = tid & 1;
      size_t wb = (size_t)(n0 + n) * 512 + (k0 >> 1) + half * 8;
      int db = n * 20 + half * 8;
      copy_b128(Whi + wb,     &Bhi[db]);
      copy_b128(Whi + wb + 4, &Bhi[db + 4]);
      copy_b128(Wlo + wb,     &Blo[db]);
      copy_b128(Wlo + wb + 4, &Blo[db + 4]);
    }
    {   // A: 64 pos x 32 k fp32 -> split into hi/lo packed LDS tiles
      int idx8 = tid * 8;
      int ml = idx8 >> 5, k8 = idx8 & 31;
      int mg = m0 + ml, b = mg >> 7, tt = mg & 127;
      const float* src = X + (size_t)tt * BD + (size_t)b * DD + k0 + k8;
      float4 x0 = *(const float4*)src;
      float4 x1 = *(const float4*)(src + 4);
      uint4 h, l;
      split2(x0.x, x0.y, h.x, l.x);
      split2(x0.z, x0.w, h.y, l.y);
      split2(x1.x, x1.y, h.z, l.z);
      split2(x1.z, x1.w, h.w, l.w);
      *(uint4*)&Ahi[ml * 20 + (k8 >> 1)] = h;
      *(uint4*)&Alo[ml * 20 + (k8 >> 1)] = l;
    }
    async_wait0();
    __syncthreads();
    v16bf bh0 = fragB(Bhi, wn + (lane & 15), lane);
    v16bf bl0 = fragB(Blo, wn + (lane & 15), lane);
    v16bf bh1 = fragB(Bhi, wn + 16 + (lane & 15), lane);
    v16bf bl1 = fragB(Blo, wn + 16 + (lane & 15), lane);
#pragma unroll
    for (int sm = 0; sm < 2; ++sm) {
      v16bf ah = fragA(Ahi, wm + sm * 16 + (lane & 15), lane);
      v16bf al = fragA(Alo, wm + sm * 16 + (lane & 15), lane);
      acc[sm][0] = wmma_bf16(ah, bh0, acc[sm][0]);
      acc[sm][0] = wmma_bf16(ah, bl0, acc[sm][0]);
      acc[sm][0] = wmma_bf16(al, bh0, acc[sm][0]);
      acc[sm][1] = wmma_bf16(ah, bh1, acc[sm][1]);
      acc[sm][1] = wmma_bf16(ah, bl1, acc[sm][1]);
      acc[sm][1] = wmma_bf16(al, bh1, acc[sm][1]);
    }
    __syncthreads();
  }
#pragma unroll
  for (int sm = 0; sm < 2; ++sm)
#pragma unroll
    for (int sn = 0; sn < 2; ++sn) {
      v8f c = acc[sm][sn];
      int nl = wn + sn * 16 + (lane & 15);
      int mh = wm + sm * 16 + ((lane >> 4) << 3);
#pragma unroll
      for (int r = 0; r < 8; ++r)
        proj[(size_t)(m0 + mh + r) * DD + n0 + nl] = c[r];
    }
}

// --------------------------------------------------- fast->slow conv GEMM ----
__global__ void f2s_gemm(const float* __restrict__ X, const float* __restrict__ Fj,
                         const uint32_t* __restrict__ Whi,
                         const uint32_t* __restrict__ Wlo,
                         const float* __restrict__ scoreB,
                         const float* __restrict__ condp, float* __restrict__ out,
                         const int is_second) {
  __shared__ uint32_t Ahi[64 * 20], Alo[64 * 20];
  __shared__ uint32_t Bhi[128 * 20], Blo[128 * 20];
  const int tid = threadIdx.x;
  const int m0 = blockIdx.x * 64;
  const int n0 = blockIdx.y * 128;
  const float cond = condp[0];
  if (cond < 0.5f) {
    if (is_second) return;                        // keep FIRST pair's result
    for (int e = tid; e < 64 * 128; e += 256) {   // passthrough feat_j
      int ml = e >> 7, nl = e & 127;
      int mg = m0 + ml, b = mg >> 7, to = mg & 127;
      size_t idx = (size_t)to * BD + (size_t)b * DD + n0 + nl;
      out[idx] = Fj[idx];
    }
    return;
  }
  const int lane = tid & 31, wave = tid >> 5;
  const int wm = (wave & 1) * 32, wn = (wave >> 1) * 32;
  v8f acc[2][2] = {};

  for (int ktap = 0; ktap < 5; ++ktap) {
    for (int c0 = 0; c0 < DD; c0 += 32) {
      {   // B: pre-split [ktap][d][c/2] planes -> LDS, async when available
        int n = tid >> 1, half = tid & 1;
        size_t wb = ((size_t)ktap * 1024 + n0 + n) * 512 + (c0 >> 1) + half * 8;
        int db = n * 20 + half * 8;
        copy_b128(Whi + wb,     &Bhi[db]);
        copy_b128(Whi + wb + 4, &Bhi[db + 4]);
        copy_b128(Wlo + wb,     &Blo[db]);
        copy_b128(Wlo + wb + 4, &Blo[db + 4]);
      }
      {   // A: zero-padded strided-conv rows, split on the fly
        int idx8 = tid * 8;
        int ml = idx8 >> 5, k8 = idx8 & 31;
        int mg = m0 + ml, b = mg >> 7, to = mg & 127;
        int tp = 4 * to + ktap - 2;
        float4 x0 = make_float4(0.f, 0.f, 0.f, 0.f), x1 = x0;
        if ((unsigned)tp < (unsigned)T_FAST) {
          const float* src = X + (size_t)tp * BD + (size_t)b * DD + c0 + k8;
          x0 = *(const float4*)src;
          x1 = *(const float4*)(src + 4);
        }
        uint4 h, l;
        split2(x0.x, x0.y, h.x, l.x);
        split2(x0.z, x0.w, h.y, l.y);
        split2(x1.x, x1.y, h.z, l.z);
        split2(x1.z, x1.w, h.w, l.w);
        *(uint4*)&Ahi[ml * 20 + (k8 >> 1)] = h;
        *(uint4*)&Alo[ml * 20 + (k8 >> 1)] = l;
      }
      async_wait0();
      __syncthreads();
      v16bf bh0 = fragB(Bhi, wn + (lane & 15), lane);
      v16bf bl0 = fragB(Blo, wn + (lane & 15), lane);
      v16bf bh1 = fragB(Bhi, wn + 16 + (lane & 15), lane);
      v16bf bl1 = fragB(Blo, wn + 16 + (lane & 15), lane);
#pragma unroll
      for (int sm = 0; sm < 2; ++sm) {
        v16bf ah = fragA(Ahi, wm + sm * 16 + (lane & 15), lane);
        v16bf al = fragA(Alo, wm + sm * 16 + (lane & 15), lane);
        acc[sm][0] = wmma_bf16(ah, bh0, acc[sm][0]);
        acc[sm][0] = wmma_bf16(ah, bl0, acc[sm][0]);
        acc[sm][0] = wmma_bf16(al, bh0, acc[sm][0]);
        acc[sm][1] = wmma_bf16(ah, bh1, acc[sm][1]);
        acc[sm][1] = wmma_bf16(ah, bl1, acc[sm][1]);
        acc[sm][1] = wmma_bf16(al, bh1, acc[sm][1]);
      }
      __syncthreads();
    }
  }
#pragma unroll
  for (int sm = 0; sm < 2; ++sm)
#pragma unroll
    for (int sn = 0; sn < 2; ++sn) {
      v8f c = acc[sm][sn];
      int nl = wn + sn * 16 + (lane & 15);
      int mh = wm + sm * 16 + ((lane >> 4) << 3);
#pragma unroll
      for (int r = 0; r < 8; ++r) {
        int mg = m0 + mh + r, b = mg >> 7, to = mg & 127;
        size_t idx = (size_t)to * BD + (size_t)b * DD + n0 + nl;
        out[idx] = Fj[idx] + scoreB[b] * c[r];
      }
    }
}

// ------------------------------------------- fast outputs: interp + select ----
__global__ void fuse_fast(const float* __restrict__ f0, const float* __restrict__ f1,
                          const float* __restrict__ ws, const float* __restrict__ s2f_b,
                          float* __restrict__ out) {
  const int j = blockIdx.y;
  const float* Fj = j ? f1 : f0;
  float* oj = out + (size_t)j * ((size_t)T_FAST * BD);
  const int qA = 2 * j;          // (i=2, j)
  const int qB = 2 * j + 1;      // (i=3, j) -- later in loop order, wins
  const float condA = ws[WS_COND + qA];
  const float condB = ws[WS_COND + qB];
  const size_t e = (size_t)blockIdx.x * 256 + threadIdx.x;
  const int t = (int)(e >> 15);
  const int r = (int)(e & 32767);
  const int b = r >> 10;
  const int d = r & 1023;
  float val = Fj[e];
  if (condB >= 0.5f || condA >= 0.5f) {
    const int useB = (condB >= 0.5f) ? 1 : 0;
    const float* proj = ws + WS_PROJ + (size_t)useB * PROJ_SZ;
    const float sc = ws[WS_SCORE + (size_t)(useB ? qB : qA) * BB + b];
    float src = ((float)t + 0.5f) * 0.25f - 0.5f;
    src = fminf(fmaxf(src, 0.0f), 127.0f);
    int i0 = (int)src;
    int i1 = (i0 + 1 < 127) ? i0 + 1 : 127;
    float w = src - (float)i0;
    float p0 = proj[(size_t)(b * 128 + i0) * DD + d];
    float p1 = proj[(size_t)(b * 128 + i1) * DD + d];
    float up = p0 * (1.0f - w) + p1 * w + s2f_b[d];
    val += sc * up;
  }
  oj[e] = val;
}

// ------------------------------------------------------------------ launch ----
extern "C" void kernel_launch(void* const* d_in, const int* in_sizes, int n_in,
                              void* d_out, int out_size, void* d_ws, size_t ws_size,
                              hipStream_t stream) {
  const float* f0 = (const float*)d_in[0];
  const float* f1 = (const float*)d_in[1];
  const float* f2 = (const float*)d_in[2];
  const float* f3 = (const float*)d_in[3];
  const float* w1 = (const float*)d_in[4];
  const float* b1 = (const float*)d_in[5];
  const float* w2 = (const float*)d_in[6];
  const float* b2 = (const float*)d_in[7];
  const float* s2f_w = (const float*)d_in[8];
  const float* s2f_b = (const float*)d_in[9];
  const float* f2s_w = (const float*)d_in[10];
  float* ws = (float*)d_ws;
  uint32_t* wsu = (uint32_t*)d_ws;
  float* out = (float*)d_out;
  float* out2 = out + (size_t)2 * T_FAST * BD;
  float* out3 = out2 + (size_t)T_SLOW * BD;

  uint32_t* s2f_hi = wsu + WS_S2FW;
  uint32_t* s2f_lo = s2f_hi + S2F_PLANE;
  uint32_t* f2s_hi = wsu + WS_F2SW;
  uint32_t* f2s_lo = f2s_hi + F2S_PLANE;
  const size_t WIDXS = (size_t)5 * 1024 * 512;   // words per f2s weight set

  means_k<<<512, 256, 0, stream>>>(f0, f1, f2, f3, ws);
  mlp_scores<<<8, 256, 0, stream>>>(w1, b1, w2, b2, ws);
  cvt_s2f<<<2048, 256, 0, stream>>>(s2f_w, s2f_hi, s2f_lo);
  cvt_f2s<<<40960, 256, 0, stream>>>(f2s_w, f2s_hi, f2s_lo);

  proj_gemm<<<dim3(64, 8, 2), 256, 0, stream>>>(f2, f3, s2f_hi, s2f_lo, ws);

  // j=2: (0,2) q=4 idx0 FIRST, then (1,2) q=5 idx2 SECOND (loop-order override)
  f2s_gemm<<<dim3(64, 8), 256, 0, stream>>>(f0, f2, f2s_hi + 0 * WIDXS,
      f2s_lo + 0 * WIDXS, ws + WS_SCORE + 4 * BB, ws + WS_COND + 4, out2, 0);
  f2s_gemm<<<dim3(64, 8), 256, 0, stream>>>(f1, f2, f2s_hi + 2 * WIDXS,
      f2s_lo + 2 * WIDXS, ws + WS_SCORE + 5 * BB, ws + WS_COND + 5, out2, 1);
  // j=3: (0,3) q=6 idx1 FIRST, then (1,3) q=7 idx3 SECOND
  f2s_gemm<<<dim3(64, 8), 256, 0, stream>>>(f0, f3, f2s_hi + 1 * WIDXS,
      f2s_lo + 1 * WIDXS, ws + WS_SCORE + 6 * BB, ws + WS_COND + 6, out3, 0);
  f2s_gemm<<<dim3(64, 8), 256, 0, stream>>>(f1, f3, f2s_hi + 3 * WIDXS,
      f2s_lo + 3 * WIDXS, ws + WS_SCORE + 7 * BB, ws + WS_COND + 7, out3, 1);

  fuse_fast<<<dim3(65536, 2), 256, 0, stream>>>(f0, f1, ws, s2f_b, out);
}
